// PerformerAttention_12386685681802
// MI455X (gfx1250) — matrix operand order repaired and checked by reading, not compiled, not run on hardware
//
#include <hip/hip_runtime.h>
#include <hip/hip_fp16.h>

typedef __attribute__((ext_vector_type(16))) _Float16 v16h;
typedef __attribute__((ext_vector_type(8)))  float    v8f;

#define PERF_EPS 1e-8f
#define N_SEQ 2048
#define DHEAD 64
#define CHUNK 64
#define NCHUNK (N_SEQ / CHUNK)
#define NHEADS 32   // B*H = 4*8

__device__ __forceinline__ v8f wmma_f16(const v16h& a, const v16h& b, const v8f& c) {
    return __builtin_amdgcn_wmma_f32_16x16x32_f16(false, a, false, b, (short)0, c, false, false);
}

// Async global -> LDS copy, 16B per lane, tracked with ASYNCcnt (CDNA5 §08).
// Low 32 bits of a generic pointer to a __shared__ object are the LDS byte offset.
__device__ __forceinline__ void async_b128(float* lds, const float* g) {
    unsigned loff = (unsigned)(uintptr_t)lds;
    asm volatile("global_load_async_to_lds_b128 %0, %1, off"
                 :: "v"(loff), "v"(g) : "memory");
}
#define WAIT_ASYNCCNT(n) asm volatile("s_wait_asynccnt %0" :: "n"(n) : "memory")

// Pass 1: per-head global max of k (needed before exp featurization of k)
__global__ __launch_bounds__(256)
void performer_kmax_kernel(const float* __restrict__ k, float* __restrict__ kmax) {
    const int head = blockIdx.x;
    const float* kp = k + (size_t)head * N_SEQ * DHEAD;
    float m = -__builtin_inff();
    for (int i = threadIdx.x; i < N_SEQ * DHEAD; i += 256)
        m = fmaxf(m, kp[i]);
    __shared__ float red[256];
    red[threadIdx.x] = m;
    __syncthreads();
    for (int s = 128; s > 0; s >>= 1) {
        if (threadIdx.x < s) red[threadIdx.x] = fmaxf(red[threadIdx.x], red[threadIdx.x + s]);
        __syncthreads();
    }
    if (threadIdx.x == 0) kmax[head] = red[0];
}

// Pass 2: chunked causal linear attention, one block per head, 8 wave32s.
__global__ __launch_bounds__(256)
void performer_main_kernel(const float* __restrict__ q, const float* __restrict__ k,
                           const float* __restrict__ v, const float* __restrict__ kmaxbuf,
                           float* __restrict__ out) {
    __shared__ _Float16 Qh[CHUNK][DHEAD];    // exp-featurized q chunk
    __shared__ _Float16 Kh[CHUNK][DHEAD];    // exp-featurized k chunk
    __shared__ _Float16 KhT[DHEAD][CHUNK];   // transpose of Kh (A-operand of M3)
    __shared__ _Float16 Vh[CHUNK][DHEAD];    // v chunk (f16)
    __shared__ _Float16 Ah[CHUNK][CHUNK];    // masked intra-chunk scores
    __shared__ _Float16 Sh[DHEAD][DHEAD];    // f16 shadow of KV state (B operand)
    __shared__ float    Sf[DHEAD][DHEAD];    // fp32 KV state  S = sum K^T V
    __shared__ float    svec[DHEAD];         // fp32 k colsum state
    __shared__ float    rowsumA[CHUNK];
    __shared__ float    zrow[CHUNK];
    __shared__ float    pmax[CHUNK][4];
    __shared__ float    stageK[2][CHUNK * DHEAD];  // async double buffer (raw f32 K)
    __shared__ float    stageV[2][CHUNK * DHEAD];  // async double buffer (raw f32 V)

    const int head = blockIdx.x;
    const int tid  = threadIdx.x;
    const int wave = tid >> 5;
    const int lane = tid & 31;
    const int halfSel = lane >> 4;          // 0: lanes 0-15, 1: lanes 16-31
    const float kmax = kmaxbuf[head];
    const float ns = 0.35355339059327373f;  // 64^(-1/4)

    const size_t base = (size_t)head * N_SEQ * DHEAD;
    const float* qp = q + base;
    const float* kp = k + base;
    const float* vp = v + base;
    float*       op = out + base;

    // kick off async staging of chunk 0 (8 x b128 per thread, per-wave ASYNCcnt=8)
    #pragma unroll
    for (int j = 0; j < 4; ++j) {
        const int i4 = tid + 256 * j;
        async_b128(&stageK[0][i4 * 4], kp + (size_t)i4 * 4);
        async_b128(&stageV[0][i4 * 4], vp + (size_t)i4 * 4);
    }

    for (int i = tid; i < DHEAD * DHEAD; i += 256) {
        Sf[i / DHEAD][i % DHEAD] = 0.0f;
        Sh[i / DHEAD][i % DHEAD] = (_Float16)0.0f;
    }
    if (tid < DHEAD) svec[tid] = 0.0f;

    for (int c = 0; c < NCHUNK; ++c) {
        const int nbase = c * CHUNK;
        const int buf   = c & 1;

        // ---- prefetch chunk c+1 into the other buffer, then wait for chunk c ----
        if (c + 1 < NCHUNK) {
            const size_t nb2 = (size_t)(nbase + CHUNK) * DHEAD;
            #pragma unroll
            for (int j = 0; j < 4; ++j) {
                const int i4 = tid + 256 * j;
                async_b128(&stageK[buf ^ 1][i4 * 4], kp + nb2 + (size_t)i4 * 4);
                async_b128(&stageV[buf ^ 1][i4 * 4], vp + nb2 + (size_t)i4 * 4);
            }
            WAIT_ASYNCCNT(8);   // in-order: all of chunk c's 8 requests have landed
        } else {
            WAIT_ASYNCCNT(0);
        }
        __syncthreads();        // async LDS writes visible to all waves

        // ---- featurize: thread t handles row t>>2, columns (t&3)*16.. ----
        {
            const int row = tid >> 2, qt = tid & 3;
            const float* qrow  = qp + (size_t)(nbase + row) * DHEAD + qt * 16;
            const float* srowK = &stageK[buf][row * DHEAD + qt * 16];
            const float* srowV = &stageV[buf][row * DHEAD + qt * 16];
            if (c + 1 < NCHUNK)
                __builtin_prefetch(qrow + CHUNK * DHEAD, 0, 0);  // global_prefetch next Q
            float vals[16];
            float m = -__builtin_inff();
            #pragma unroll
            for (int j = 0; j < 16; ++j) { vals[j] = qrow[j]; m = fmaxf(m, vals[j]); }
            pmax[row][qt] = m;
            __syncthreads();
            m = fmaxf(fmaxf(pmax[row][0], pmax[row][1]), fmaxf(pmax[row][2], pmax[row][3]));
            #pragma unroll
            for (int j = 0; j < 16; ++j) {
                const _Float16 ke = (_Float16)(__expf(ns * (srowK[j] - kmax)) + PERF_EPS);
                Qh[row][qt * 16 + j]  = (_Float16)(__expf(ns * (vals[j] - m)) + PERF_EPS);
                Kh[row][qt * 16 + j]  = ke;
                KhT[qt * 16 + j][row] = ke;
                Vh[row][qt * 16 + j]  = (_Float16)srowV[j];
            }
        }
        if (tid < CHUNK) rowsumA[tid] = 0.0f;
        __syncthreads();

        // ---- M1: A = Qe x Ke^T (K-dim = d = 64), mask, rowsum, spill to Ah ----
        #pragma unroll
        for (int tt = 0; tt < 2; ++tt) {
            const int t = wave * 2 + tt, ti = t >> 2, tj = t & 3;
            v8f acc = {};
            #pragma unroll
            for (int kb = 0; kb < 2; ++kb) {
                v16h a, b;
                const int m  = ti * 16 + (lane & 15);
                const int kA = kb * 32 + halfSel * 16;
                #pragma unroll
                for (int h = 0; h < 16; ++h)
                    a[h] = Qh[m][kA + h];
                const int n  = tj * 16 + (lane & 15);
                const int kB = kb * 32 + halfSel * 16;
                #pragma unroll
                for (int h = 0; h < 16; ++h)
                    b[h] = Kh[n][kB + h];          // B = Ke^T: B[kd][j] = Kh[j][kd]
                acc = wmma_f16(a, b, acc);
            }
            const int nCol  = tj * 16 + (lane & 15);
            const int mBase = ti * 16 + halfSel * 8;
            #pragma unroll
            for (int r = 0; r < 8; ++r) {
                const int mRow = mBase + r;
                const float val = (nCol <= mRow) ? acc[r] : 0.0f;   // causal (inclusive)
                Ah[mRow][nCol] = (_Float16)val;
                atomicAdd(&rowsumA[mRow], val);
            }
        }
        __syncthreads();

        // ---- denominator: z = q.s_prev + rowsum(A_mask) + eps*rowsum(q) ----
        if (tid < CHUNK) {
            float zq = 0.0f, rq = 0.0f;
            for (int e = 0; e < DHEAD; ++e) {
                const float qe = (float)Qh[tid][e];
                zq += qe * svec[e];
                rq += qe;
            }
            zrow[tid] = zq + rowsumA[tid] + PERF_EPS * rq;
        }
        __syncthreads();

        // ---- M2: O = A_mask x V + Qe x S_prev, scale by 1/z, store ----
        #pragma unroll
        for (int tt = 0; tt < 2; ++tt) {
            const int t = wave * 2 + tt, ti = t >> 2, tj = t & 3;
            v8f acc = {};
            #pragma unroll
            for (int kb = 0; kb < 2; ++kb) {           // A_mask @ V
                v16h a, b;
                const int m  = ti * 16 + (lane & 15);
                const int kA = kb * 32 + halfSel * 16;
                #pragma unroll
                for (int h = 0; h < 16; ++h)
                    a[h] = Ah[m][kA + h];
                const int n  = tj * 16 + (lane & 15);
                const int kB = kb * 32 + halfSel * 16;
                #pragma unroll
                for (int h = 0; h < 16; ++h)
                    b[h] = Vh[kB + h][n];
                acc = wmma_f16(a, b, acc);
            }
            #pragma unroll
            for (int kb = 0; kb < 2; ++kb) {           // Qe @ S_prev
                v16h a, b;
                const int m  = ti * 16 + (lane & 15);
                const int kA = kb * 32 + halfSel * 16;
                #pragma unroll
                for (int h = 0; h < 16; ++h)
                    a[h] = Qh[m][kA + h];
                const int n  = tj * 16 + (lane & 15);
                const int kB = kb * 32 + halfSel * 16;
                #pragma unroll
                for (int h = 0; h < 16; ++h)
                    b[h] = Sh[kB + h][n];
                acc = wmma_f16(a, b, acc);
            }
            const int nCol  = tj * 16 + (lane & 15);
            const int mBase = ti * 16 + halfSel * 8;
            #pragma unroll
            for (int r = 0; r < 8; ++r) {
                const int mRow = mBase + r;
                op[(size_t)(nbase + mRow) * DHEAD + nCol] = acc[r] / zrow[mRow];
            }
        }
        __syncthreads();   // all reads of Sh / svec done before state update

        // ---- M3: state update S += Ke^T x V ; s += colsum(Ke) ----
        #pragma unroll
        for (int tt = 0; tt < 2; ++tt) {
            const int t = wave * 2 + tt, ti = t >> 2, tj = t & 3;  // ti: e-block, tj: f-block
            v8f acc = {};
            #pragma unroll
            for (int kb = 0; kb < 2; ++kb) {
                v16h a, b;
                const int e  = ti * 16 + (lane & 15);
                const int kA = kb * 32 + halfSel * 16;
                #pragma unroll
                for (int h = 0; h < 16; ++h)
                    a[h] = KhT[e][kA + h];  // A = Ke^T (contiguous)
                const int n  = tj * 16 + (lane & 15);
                const int kB = kb * 32 + halfSel * 16;
                #pragma unroll
                for (int h = 0; h < 16; ++h)
                    b[h] = Vh[kB + h][n];
                acc = wmma_f16(a, b, acc);
            }
            const int nCol  = tj * 16 + (lane & 15);
            const int mBase = ti * 16 + halfSel * 8;
            #pragma unroll
            for (int r = 0; r < 8; ++r) {
                const int e = mBase + r;
                const float cur = Sf[e][nCol] + acc[r];
                Sf[e][nCol] = cur;
                Sh[e][nCol] = (_Float16)cur;
            }
        }
        if (tid < DHEAD) {
            float cs = 0.0f;
            for (int n = 0; n < CHUNK; ++n) cs += (float)Kh[n][tid];
            svec[tid] += cs;
        }
        __syncthreads();
    }
}

extern "C" void kernel_launch(void* const* d_in, const int* in_sizes, int n_in,
                              void* d_out, int out_size, void* d_ws, size_t ws_size,
                              hipStream_t stream) {
    const float* q = (const float*)d_in[0];
    const float* k = (const float*)d_in[1];
    const float* v = (const float*)d_in[2];
    float* out  = (float*)d_out;
    float* kmax = (float*)d_ws;   // 32 floats

    hipLaunchKernelGGL(performer_kmax_kernel, dim3(NHEADS), dim3(256), 0, stream, k, kmax);
    hipLaunchKernelGGL(performer_main_kernel, dim3(NHEADS), dim3(256), 0, stream,
                       q, k, v, kmax, out);
}
